// TripletLoss_55877524521543
// MI455X (gfx1250) — matrix operand order repaired and checked by reading, not compiled, not run on hardware
//
#include <hip/hip_runtime.h>
#include <cstdint>
#include <cstddef>

// ---------------------------------------------------------------------------
// Types for WMMA (gfx1250, wave32)
// ---------------------------------------------------------------------------
typedef __attribute__((ext_vector_type(16))) __bf16 v16bf;
typedef __attribute__((ext_vector_type(8)))  __bf16 v8bf;
typedef __attribute__((ext_vector_type(4)))  __bf16 v4bf;
typedef __attribute__((ext_vector_type(8)))  float  v8f;
typedef __attribute__((ext_vector_type(4)))  float  v4f;

union V16 { v16bf v; v8bf h[2]; };

#define NB       32768
#define ND       512
#define NC       200
#define NCPAD    208      // 13 tiles of 16 classes
#define NCTILES  13
#define CHUNKS   64       // row chunks for deterministic segment sums
#define CHSZ     (NB / CHUNKS)   // 512
#define NTILES   (NB / 16)       // 2048 row tiles
#define BIGF     3.0e38f

// ---------------------------------------------------------------------------
// Kernel 1a: per-(class, row-chunk) partial feature sums (deterministic order)
// grid = NC*CHUNKS blocks, 256 threads. Each thread owns dims t and t+256.
// ---------------------------------------------------------------------------
__global__ void k_protopart(const float* __restrict__ feat,
                            const int*   __restrict__ labels,
                            float* __restrict__ ps, int* __restrict__ pc) {
  const int c     = blockIdx.x >> 6;       // 0..199
  const int chunk = blockIdx.x & 63;       // 0..63
  const int t     = threadIdx.x;
  const int i0    = chunk * CHSZ;
  float s0 = 0.f, s1 = 0.f;
  int cnt = 0;
  for (int i = i0; i < i0 + CHSZ; ++i) {   // uniform scalar label scan
    if (labels[i] == c) {
      const float* fr = feat + (size_t)i * ND;
      s0 += fr[t];
      s1 += fr[t + 256];
      ++cnt;
    }
  }
  float* o = ps + (size_t)blockIdx.x * ND;
  o[t]       = s0;
  o[t + 256] = s1;
  if (t == 0) pc[blockIdx.x] = cnt;
}

// ---------------------------------------------------------------------------
// Kernel 1b: finalize prototypes (f32 + bf16 padded) and p2 = |proto|^2
// grid = NCPAD blocks, 256 threads.
// ---------------------------------------------------------------------------
__global__ void k_protofin(const float* __restrict__ ps, const int* __restrict__ pc,
                           float* __restrict__ protoF, __bf16* __restrict__ protoBf,
                           float* __restrict__ p2) {
  const int c = blockIdx.x;
  const int t = threadIdx.x;
  __shared__ float red[256];
  if (c >= NC) {                      // padded classes: zero bf16 rows
    protoBf[(size_t)c * ND + t]       = (__bf16)0.f;
    protoBf[(size_t)c * ND + t + 256] = (__bf16)0.f;
    if (t == 0) p2[c] = 0.f;
    return;
  }
  float s0 = 0.f, s1 = 0.f;
  int cnt = 0;
  for (int k = 0; k < CHUNKS; ++k) {  // fixed chunk order -> deterministic
    const float* o = ps + (size_t)(c * CHUNKS + k) * ND;
    s0 += o[t];
    s1 += o[t + 256];
    cnt += pc[c * CHUNKS + k];
  }
  const float inv = 1.f / ((float)cnt + 1e-8f);
  const float p0 = s0 * inv, p1 = s1 * inv;
  protoF[(size_t)c * ND + t]        = p0;
  protoF[(size_t)c * ND + t + 256]  = p1;
  protoBf[(size_t)c * ND + t]       = (__bf16)p0;
  protoBf[(size_t)c * ND + t + 256] = (__bf16)p1;
  red[t] = p0 * p0 + p1 * p1;
  __syncthreads();
  for (int o = 128; o > 0; o >>= 1) {
    if (t < o) red[t] += red[t + o];
    __syncthreads();
  }
  if (t == 0) p2[c] = red[0];
}

// ---------------------------------------------------------------------------
// Kernel 2: WMMA scoring (A and B tiles staged in LDS -> no spills, no
// global-latency stalls in the MAC loop) + argmin + exact fp32 distances.
// grid = NTILES/2 blocks x 64 threads (2 waves, one 16-row tile each).
// Static LDS: 2*16KB (A) + 16KB (B) = 48KB.
// ---------------------------------------------------------------------------
__global__ __launch_bounds__(64)
void k_main(const float* __restrict__ feat,
            const int*   __restrict__ labels,
            const float* __restrict__ protoF,
            const __bf16* __restrict__ protoBf,
            const float* __restrict__ p2,
            float* __restrict__ partial) {
  __shared__ __bf16 aT[2][16 * ND];   // per-wave feature tile (bf16)
  __shared__ __bf16 bT[16 * ND];      // shared class tile (bf16)

  const int lane    = threadIdx.x & 31;
  const int wv      = threadIdx.x >> 5;
  const int tile    = blockIdx.x * 2 + wv;
  const int rowBase = tile * 16;
  const int nsub    = lane & 15;
  const int half    = lane >> 4;

  // ---- Stage A: contiguous 16x512 f32 tile -> bf16 in this wave's slab ----
  {
    const float* src = feat + (size_t)rowBase * ND;
    __bf16* dst = aT[wv];
    for (int i = lane * 8; i < 16 * ND; i += 32 * 8) {
      v4f q0 = *(const v4f*)(src + i);
      v4f q1 = *(const v4f*)(src + i + 4);
      v8bf o;
      #pragma unroll
      for (int j = 0; j < 4; ++j) { o[j] = (__bf16)q0[j]; o[4 + j] = (__bf16)q1[j]; }
      *(v8bf*)(dst + i) = o;
    }
  }

  // Labels of the 8 rows this lane's C/D elements belong to (M = v + half*8)
  int lab[8];
  #pragma unroll
  for (int v = 0; v < 8; ++v) lab[v] = labels[rowBase + half * 8 + v];

  float best[8];
  int   bidx[8];
  #pragma unroll
  for (int v = 0; v < 8; ++v) { best[v] = BIGF; bidx[v] = 0; }

  // ---- Score all 208 (padded) classes, 13 tiles of 16 ----
  for (int ct = 0; ct < NCTILES; ++ct) {
    __syncthreads();                  // protect bT reuse across iterations
    {                                 // stage B tile: 16 classes x 512 bf16
      const __bf16* src = protoBf + (size_t)ct * 16 * ND;
      for (int i = threadIdx.x * 8; i < 16 * ND; i += 64 * 8)
        *(v8bf*)(bT + i) = *(const v8bf*)(src + i);
    }
    __syncthreads();

    const int c = ct * 16 + nsub;     // this lane's class column
    v8f acc = {0.f, 0.f, 0.f, 0.f, 0.f, 0.f, 0.f, 0.f};
    // A layout (16-bit 16x32): lanes 0-15 row M=lane, K = 0..7 & 16..23;
    //                          lanes 16-31 row M=lane-16, K = 8..15 & 24..31.
    // B layout (16-bit 32x16): lanes 0-15 hold K = 0..7 & 16..23, lanes 16-31 K = 8..15 & 24..31.
    const __bf16* pa = aT[wv] + nsub * ND + half * 8;
    const __bf16* pb = bT + nsub * ND + half * 8;
    #pragma unroll
    for (int kc = 0; kc < 16; ++kc) {
      V16 av, bv;
      av.h[0] = *(const v8bf*)(pa + kc * 32);
      av.h[1] = *(const v8bf*)(pa + kc * 32 + 16);
      bv.h[0] = *(const v8bf*)(pb + kc * 32);
      bv.h[1] = *(const v8bf*)(pb + kc * 32 + 16);
      acc = __builtin_amdgcn_wmma_f32_16x16x32_bf16(
          /*neg_a=*/false, av.v, /*neg_b=*/false, bv.v,
          /*c_mod=*/(short)0, acc, /*reuse_a=*/false, /*reuse_b=*/false);
    }
    const float p2c = (c < NC) ? p2[c] : BIGF;
    #pragma unroll
    for (int v = 0; v < 8; ++v) {
      float sc = p2c - 2.0f * acc[v];             // f2 is row-constant: drop
      if (c == lab[v]) sc = BIGF;                 // mask positive class
      if (sc < best[v]) { best[v] = sc; bidx[v] = c; }
    }
  }

  // ---- Argmin butterfly across each 16-lane half (tie -> smaller class) ----
  #pragma unroll
  for (int v = 0; v < 8; ++v) {
    #pragma unroll
    for (int off = 8; off >= 1; off >>= 1) {
      float ov = __shfl_xor(best[v], off, 32);
      int   oi = __shfl_xor(bidx[v], off, 32);
      if (ov < best[v] || (ov == best[v] && oi < bidx[v])) {
        best[v] = ov; bidx[v] = oi;
      }
    }
  }

  // ---- Exact fp32 d_ap / d_an per row + loss, whole wave cooperates ----
  float lsum = 0.f;
  #pragma unroll
  for (int v = 0; v < 8; ++v) {
    const int negLo = __shfl(bidx[v], 0, 32);     // row v      (lanes 0-15)
    const int negHi = __shfl(bidx[v], 16, 32);    // row v + 8  (lanes 16-31)
    #pragma unroll
    for (int hh = 0; hh < 2; ++hh) {
      const int r  = rowBase + hh * 8 + v;
      const int lc = labels[r];
      const int nc = hh ? negHi : negLo;
      const float* fr = feat   + (size_t)r  * ND;
      const float* pp = protoF + (size_t)lc * ND;
      const float* pn = protoF + (size_t)nc * ND;
      float dap = 0.f, dan = 0.f;
      #pragma unroll
      for (int d = lane * 4; d < ND; d += 128) {  // float4 strided
        v4f fv = *(const v4f*)(fr + d);
        v4f pv = *(const v4f*)(pp + d);
        v4f nv = *(const v4f*)(pn + d);
        #pragma unroll
        for (int j = 0; j < 4; ++j) {
          const float t1 = fv[j] - pv[j]; dap += t1 * t1;
          const float t2 = fv[j] - nv[j]; dan += t2 * t2;
        }
      }
      #pragma unroll
      for (int off = 16; off >= 1; off >>= 1) {
        dap += __shfl_xor(dap, off, 32);
        dan += __shfl_xor(dan, off, 32);
      }
      // Reference's sequential normalization
      const float dapn = dap / (dap + dan + 1e-8f);
      const float dann = dan / (dapn + dan + 1e-8f);
      lsum += fmaxf(0.f, 1.5f * dapn - 0.8f * dann + 0.6f);
    }
  }
  if (lane == 0) partial[tile] = lsum;
}

// ---------------------------------------------------------------------------
// Kernel 3: deterministic final reduction -> abs(lambda * mean)
// ---------------------------------------------------------------------------
__global__ void k_reduce(const float* __restrict__ partial, float* __restrict__ out) {
  __shared__ float red[256];
  const int t = threadIdx.x;
  float s = 0.f;
  for (int i = t; i < NTILES; i += 256) s += partial[i];
  red[t] = s;
  __syncthreads();
  for (int o = 128; o > 0; o >>= 1) {
    if (t < o) red[t] += red[t + o];
    __syncthreads();
  }
  if (t == 0) out[0] = fabsf(0.2f * red[0] / (float)NB);
}

// ---------------------------------------------------------------------------
// Host launcher
// ---------------------------------------------------------------------------
extern "C" void kernel_launch(void* const* d_in, const int* in_sizes, int n_in,
                              void* d_out, int out_size, void* d_ws, size_t ws_size,
                              hipStream_t stream) {
  (void)in_sizes; (void)n_in; (void)out_size; (void)ws_size;
  const float* feat   = (const float*)d_in[0];
  const int*   labels = (const int*)d_in[1];
  float*       out    = (float*)d_out;

  char*  w   = (char*)d_ws;
  size_t off = 0;
  auto take = [&](size_t bytes) -> void* {
    void* p = w + off;
    off += (bytes + 255) & ~(size_t)255;
    return p;
  };
  float*  ps      = (float*) take((size_t)NC * CHUNKS * ND * sizeof(float)); // 26.2 MB
  int*    pc      = (int*)   take((size_t)NC * CHUNKS * sizeof(int));
  float*  protoF  = (float*) take((size_t)NC * ND * sizeof(float));
  __bf16* protoBf = (__bf16*)take((size_t)NCPAD * ND * sizeof(__bf16));
  float*  p2      = (float*) take((size_t)NCPAD * sizeof(float));
  float*  partial = (float*) take((size_t)NTILES * sizeof(float));

  k_protopart<<<NC * CHUNKS, 256, 0, stream>>>(feat, labels, ps, pc);
  k_protofin <<<NCPAD,       256, 0, stream>>>(ps, pc, protoF, protoBf, p2);
  k_main     <<<NTILES / 2,  64,  0, stream>>>(feat, labels, protoF, protoBf, p2, partial);
  k_reduce   <<<1,           256, 0, stream>>>(partial, out);
}
